// ODENet_19078244729126
// MI455X (gfx1250) — matrix operand order repaired and checked by reading, not compiled, not run on hardware
//
#include <hip/hip_runtime.h>
#include <hip/hip_bf16.h>
#include <math.h>

typedef _Float16 v16h __attribute__((ext_vector_type(16)));
typedef _Float16 v8h  __attribute__((ext_vector_type(8)));
typedef float    v8f  __attribute__((ext_vector_type(8)));

#define BATCH   8192
#define RS      136          // padded row stride (halves) of 16x128 activation tile
#define WPB     8            // waves per block
#define NBLK    64           // 64 blocks * 8 waves * 16 rows = 8192

// Dormand-Prince 5(4) tableau
constexpr float cA21 = (float)(1.0/5.0);
constexpr float cA31 = (float)(3.0/40.0),  cA32 = (float)(9.0/40.0);
constexpr float cA41 = (float)(44.0/45.0), cA42 = (float)(-56.0/15.0), cA43 = (float)(32.0/9.0);
constexpr float cA51 = (float)(19372.0/6561.0), cA52 = (float)(-25360.0/2187.0),
                cA53 = (float)(64448.0/6561.0), cA54 = (float)(-212.0/729.0);
constexpr float cA61 = (float)(9017.0/3168.0),  cA62 = (float)(-355.0/33.0),
                cA63 = (float)(46732.0/5247.0), cA64 = (float)(49.0/176.0),
                cA65 = (float)(-5103.0/18656.0);
constexpr float cB1 = (float)(35.0/384.0), cB3 = (float)(500.0/1113.0),
                cB4 = (float)(125.0/192.0), cB5 = (float)(-2187.0/6784.0), cB6 = (float)(11.0/84.0);
constexpr float cE1 = (float)(35.0/384.0 - 5179.0/57600.0);
constexpr float cE3 = (float)(500.0/1113.0 - 7571.0/16695.0);
constexpr float cE4 = (float)(125.0/192.0 - 393.0/640.0);
constexpr float cE5 = (float)(-2187.0/6784.0 + 92097.0/339200.0);
constexpr float cE6 = (float)(11.0/84.0 - 187.0/2100.0);
constexpr float cE7 = (float)(-1.0/40.0);
constexpr float cRTOL = 1e-3f, cATOL = 1e-4f;

// Branch-free ELU(alpha=1): max(v,0) + (exp(min(v,0)) - 1)
__device__ __forceinline__ float elu1(float v) {
    return fmaxf(v, 0.f) + (__expf(fminf(v, 0.f)) - 1.f);
}

// B fragment: 16 contiguous halves per lane (pre-arranged layout), two b128 loads
__device__ __forceinline__ v16h load_bfrag(const _Float16* p) {
    v8h lo = *(const v8h*)p;
    v8h hh = *(const v8h*)(p + 8);
    return __builtin_shufflevector(lo, hh, 0,1,2,3,4,5,6,7,8,9,10,11,12,13,14,15);
}

// A fragments (16x128 activation tile, 4 K-chunks of 32) in contiguous k order (k = 16*hi + j), matching the packed B
__device__ __forceinline__ void load_afrags(const _Float16* HA, int lane, v16h* A) {
    const int m = lane & 15, hi = lane >> 4;
#pragma unroll
    for (int kc = 0; kc < 4; ++kc) {
        const _Float16* rp = &HA[m * RS + kc * 32 + 16 * hi]; // K = kc*32 + 16*hi + j
        v8h lo = *(const v8h*)rp;
        v8h hh = *(const v8h*)(rp + 8);                       // K = +8
        A[kc] = __builtin_shufflevector(lo, hh, 0,1,2,3,4,5,6,7,8,9,10,11,12,13,14,15);
    }
}

// C/D epilogue: bias + ELU + f16 store into activation tile (row M = r + 8*hi)
__device__ __forceinline__ void epilogue(v8f c, int nt, const float* bias,
                                         _Float16* HA, int lane) {
    const int m = lane & 15, hi = lane >> 4;
    const int n = nt * 16 + m;
    const float bb = bias[n];
#pragma unroll
    for (int r = 0; r < 8; ++r)
        HA[(r + 8 * hi) * RS + n] = (_Float16)elu1(c[r] + bb);
}

// 128x128 hidden layer, in place on a per-wave 16x128 tile (f32 accumulate)
__device__ __forceinline__ void hidden_layer(_Float16* HA, const _Float16* WB,
                                             const float* bias, int lane)
{
    v16h A[4];
    load_afrags(HA, lane, A);
#pragma unroll
    for (int nt = 0; nt < 8; ++nt) {
        v8f c = {};
#pragma unroll
        for (int kc = 0; kc < 4; ++kc) {
            v16h B = load_bfrag(&WB[(((nt * 4 + kc) * 32) + lane) * 16]);
            c = __builtin_amdgcn_wmma_f32_16x16x32_f16(false, A[kc], false, B,
                                                       (short)0, c, false, false);
        }
        epilogue(c, nt, bias, HA, lane);
    }
}

// ---------------------------------------------------------------------------
// One DoPri5 trial step for the full batch: produces y5 + per-wave error sums.
// ---------------------------------------------------------------------------
__global__ void __launch_bounds__(256)
ode_step_kernel(const float* __restrict__ tcA, const float* __restrict__ hA,
                const float* __restrict__ tA, int p, int s,
                const float* __restrict__ ycur, float* __restrict__ y5out,
                float* __restrict__ wsum,
                const _Float16* __restrict__ WB1g, const _Float16* __restrict__ WB2g,
                const _Float16* __restrict__ WB3g, const _Float16* __restrict__ WB4g,
                const float* __restrict__ b1, const float* __restrict__ b2,
                const float* __restrict__ b3, const float* __restrict__ b4)
{
    __shared__ _Float16 sWB1[4096];          //  8 KB: 2->128 padded, B layout
    __shared__ _Float16 sWB2[16384];         // 32 KB
    __shared__ _Float16 sWB3[16384];         // 32 KB
    __shared__ _Float16 sWB4[2048];          //  4 KB: 128->2 padded, B layout
    __shared__ _Float16 sHA[WPB][16 * RS];   // per-wave activation tiles
    __shared__ float    sYS[WPB][32];        // per-wave 16x2 stage input
    __shared__ float    sOUT[WPB][32];       // per-wave 16x2 head output
    __shared__ float    sb1[128], sb2[128], sb3[128], sb4[2];

    const int tid = threadIdx.x;

    // cooperative weight load into LDS
    {
        const uint4* g1 = (const uint4*)WB1g;
        const uint4* g2 = (const uint4*)WB2g;
        const uint4* g3 = (const uint4*)WB3g;
        const uint4* g4 = (const uint4*)WB4g;
        uint4* d1 = (uint4*)sWB1;
        uint4* d2 = (uint4*)sWB2;
        uint4* d3 = (uint4*)sWB3;
        uint4* d4 = (uint4*)sWB4;
#pragma unroll
        for (int i = tid; i < 2048; i += 256) { d2[i] = g2[i]; d3[i] = g3[i]; }
        d1[tid] = g1[tid];
        d1[tid + 256] = g1[tid + 256];
        d4[tid] = g4[tid];
        if (tid < 128) { sb1[tid] = b1[tid]; sb2[tid] = b2[tid]; sb3[tid] = b3[tid]; }
        if (tid < 2) sb4[tid] = b4[tid];
    }
    __syncthreads();

    const int lane  = tid & 31;
    const int w     = tid >> 5;
    const int m     = lane & 15;
    const int hi    = lane >> 4;
    const int gwave = blockIdx.x * WPB + w;
    const int gi    = (gwave * 16 + m) * 2 + hi;   // one (row, component) per lane

    const float tc = tcA[s];
    const float hh = hA[s];
    const float t1 = tA[p + 1];
    const float remaining = t1 - tc;
    const bool  done = remaining <= 1e-9f;
    const float heff = done ? hh : fminf(hh, remaining);

    const float y = ycur[gi];
    _Float16* HA = sHA[w];
    float*    YS = sYS[w];
    float*    OUT = sOUT[w];

    auto evalf = [&](float ys) -> float {
        YS[m * 2 + hi] = ys;
        __syncthreads();
        const float y0 = YS[m * 2 + 0];
        const float y1 = YS[m * 2 + 1];
        // layer 1: 2 -> 128 via WMMA; A has only K=0,1 populated (lanes 0-15)
        {
            v16h A1 = {};
            A1[0] = (_Float16)(hi ? 0.f : y0);
            A1[1] = (_Float16)(hi ? 0.f : y1);
#pragma unroll
            for (int nt = 0; nt < 8; ++nt) {
                v16h B = load_bfrag(&sWB1[(nt * 32 + lane) * 16]);
                v8f c = {};
                c = __builtin_amdgcn_wmma_f32_16x16x32_f16(false, A1, false, B,
                                                           (short)0, c, false, false);
                epilogue(c, nt, sb1, HA, lane);
            }
        }
        // layers 2 & 3: 128x128 WMMA
        hidden_layer(HA, sWB2, sb2, lane);
        hidden_layer(HA, sWB3, sb3, lane);
        // layer 4: 128 -> 2 via WMMA (B zero-padded to 16 columns)
        {
            v16h A[4];
            load_afrags(HA, lane, A);
            v8f c = {};
#pragma unroll
            for (int kc = 0; kc < 4; ++kc) {
                v16h B = load_bfrag(&sWB4[(kc * 32 + lane) * 16]);
                c = __builtin_amdgcn_wmma_f32_16x16x32_f16(false, A[kc], false, B,
                                                           (short)0, c, false, false);
            }
            const int n = lane & 15;
            if (n < 2) {                 // lanes 0,1,16,17 own the real columns
                const float bb = sb4[n];
#pragma unroll
                for (int r = 0; r < 8; ++r)
                    OUT[(r + 8 * hi) * 2 + n] = c[r] + bb;
            }
        }
        return OUT[m * 2 + hi];          // same-wave LDS ordering via DScnt
    };

    const float k1 = evalf(y);
    const float k2 = evalf(y + heff * (cA21 * k1));
    const float k3 = evalf(y + heff * (cA31 * k1 + cA32 * k2));
    const float k4 = evalf(y + heff * (cA41 * k1 + cA42 * k2 + cA43 * k3));
    const float k5 = evalf(y + heff * (cA51 * k1 + cA52 * k2 + cA53 * k3 + cA54 * k4));
    const float k6 = evalf(y + heff * (cA61 * k1 + cA62 * k2 + cA63 * k3 + cA64 * k4 + cA65 * k5));
    const float y5v = y + heff * (cB1 * k1 + cB3 * k3 + cB4 * k4 + cB5 * k5 + cB6 * k6);
    const float k7 = evalf(y5v);   // FSAL stage, error estimate only

    const float err = heff * (cE1 * k1 + cE3 * k3 + cE4 * k4 + cE5 * k5 + cE6 * k6 + cE7 * k7);
    const float tol = cATOL + cRTOL * fmaxf(fabsf(y), fabsf(y5v));
    float r = err / tol;
    r = r * r;
#pragma unroll
    for (int off = 16; off >= 1; off >>= 1) r += __shfl_xor(r, off, 32);
    if (lane == 0) wsum[gwave] = r;        // deterministic partial (no atomics)
    y5out[gi] = y5v;
}

// ---------------------------------------------------------------------------
// Global accept/reject + scalar-state advance (step-indexed to avoid hazards).
// ---------------------------------------------------------------------------
__global__ void __launch_bounds__(256)
ode_commit_kernel(float* __restrict__ tcA, float* __restrict__ hA,
                  const float* __restrict__ tA, int p, int s, int isEnd,
                  float* __restrict__ ycur, const float* __restrict__ y5,
                  const float* __restrict__ wsum, float* __restrict__ dout)
{
    __shared__ float red[256];
    const int tid = threadIdx.x;
    red[tid] = wsum[tid] + wsum[tid + 256];
    __syncthreads();
#pragma unroll
    for (int off = 128; off >= 1; off >>= 1) {
        if (tid < off) red[tid] += red[tid + off];
        __syncthreads();
    }
    const float errsum = red[0];                    // identical in every block

    const float tc = tcA[s];
    const float hh = hA[s];
    const float t1 = tA[p + 1];
    const float remaining = t1 - tc;
    const bool  done = remaining <= 1e-9f;
    const float heff = done ? hh : fminf(hh, remaining);
    const float err_norm = sqrtf(errsum * (1.0f / (float)(BATCH * 2)));
    const bool  accept = (err_norm <= 1.0f) && !done;

    const int i = blockIdx.x * 256 + tid;
    const float yn = accept ? y5[i] : ycur[i];
    ycur[i] = yn;
    if (isEnd) dout[(p + 1) * (BATCH * 2) + i] = yn;

    if (blockIdx.x == 0 && tid == 0) {
        float factor = 0.9f * powf(fmaxf(err_norm, 1e-10f), -0.2f);
        factor = fminf(fmaxf(factor, 0.2f), 10.0f);
        hA[s + 1]  = done ? hh : heff * factor;
        tcA[s + 1] = isEnd ? t1 : (accept ? tc + heff : tc);
    }
}

// ---------------------------------------------------------------------------
// Rearrange weights into f16 WMMA-B fragment order.
//   WB2/WB3: [nt][kc][lane][j] = W[kc*32 + 16*(lane>>4) + j][nt*16 + (lane&15)]
//   WB1:     [nt][lane][j]     = (lane<16 && j<2) ? W1[j][nt*16+(lane&15)] : 0
//   WB4:     [kc][lane][j]     = (n<2) ? W4[K][n] : 0,  n = lane&15
// ---------------------------------------------------------------------------
__global__ void __launch_bounds__(256)
ode_prep_kernel(const float* __restrict__ W1, const float* __restrict__ W2,
                const float* __restrict__ W3, const float* __restrict__ W4,
                _Float16* __restrict__ WB1, _Float16* __restrict__ WB2,
                _Float16* __restrict__ WB3, _Float16* __restrict__ WB4)
{
    const int tid = blockIdx.x * 256 + threadIdx.x;
    if (tid < 32768) {
        const int u    = tid & 16383;
        const int j    = u & 15;
        const int lane = (u >> 4) & 31;
        const int kc   = (u >> 9) & 3;
        const int nt   = u >> 11;
        const int k = kc * 32 + 16 * (lane >> 4) + j;
        const int n = nt * 16 + (lane & 15);
        if (tid < 16384) WB2[u] = (_Float16)W2[k * 128 + n];
        else             WB3[u] = (_Float16)W3[k * 128 + n];
    } else if (tid < 36864) {
        const int u    = tid - 32768;      // 0..4095
        const int j    = u & 15;
        const int lane = (u >> 4) & 31;
        const int nt   = u >> 9;
        const float v = (lane < 16 && j < 2) ? W1[j * 128 + nt * 16 + (lane & 15)] : 0.f;
        WB1[u] = (_Float16)v;
    } else if (tid < 38912) {
        const int u    = tid - 36864;      // 0..2047
        const int j    = u & 15;
        const int lane = (u >> 4) & 31;
        const int kc   = u >> 9;
        const int k = kc * 32 + 16 * (lane >> 4) + j;
        const int n = lane & 15;
        const float v = (n < 2) ? W4[k * 2 + n] : 0.f;
        WB4[u] = (_Float16)v;
    }
}

__global__ void __launch_bounds__(256)
ode_init_kernel(const float* __restrict__ x0, const float* __restrict__ tA,
                float* __restrict__ ycur, float* __restrict__ dout,
                float* __restrict__ tcA, float* __restrict__ hA)
{
    const int i = blockIdx.x * 256 + threadIdx.x;     // 0..16383
    const float v = x0[i];
    ycur[i] = v;
    dout[i] = v;                                       // output[0] = x0
    if (i == 0) { tcA[0] = tA[0]; hA[0] = (tA[1] - tA[0]) * 0.5f; }
}

// ---------------------------------------------------------------------------
extern "C" void kernel_launch(void* const* d_in, const int* in_sizes, int n_in,
                              void* d_out, int out_size, void* d_ws, size_t ws_size,
                              hipStream_t stream)
{
    (void)in_sizes; (void)n_in; (void)out_size; (void)ws_size;
    const float* x0 = (const float*)d_in[0];
    const float* t  = (const float*)d_in[1];
    const float* W1 = (const float*)d_in[2];
    const float* b1 = (const float*)d_in[3];
    const float* W2 = (const float*)d_in[4];
    const float* b2 = (const float*)d_in[5];
    const float* W3 = (const float*)d_in[6];
    const float* b3 = (const float*)d_in[7];
    const float* W4 = (const float*)d_in[8];
    const float* b4 = (const float*)d_in[9];
    float* out = (float*)d_out;

    float* ws   = (float*)d_ws;
    float* ycur = ws;                        // 16384 f32
    float* y5   = ws + 16384;                // 16384 f32
    float* wsum = ws + 32768;                // 512   f32
    float* tcA  = ws + 33280;                // 57    f32
    float* hA   = ws + 33344;                // 57    f32
    _Float16* WB2 = (_Float16*)(ws + 33408); // 16384 f16
    _Float16* WB3 = (_Float16*)(ws + 41600); // 16384 f16
    _Float16* WB1 = (_Float16*)(ws + 49792); //  4096 f16
    _Float16* WB4 = (_Float16*)(ws + 51840); //  2048 f16

    ode_prep_kernel<<<152, 256, 0, stream>>>(W1, W2, W3, W4, WB1, WB2, WB3, WB4);
    ode_init_kernel<<<64, 256, 0, stream>>>(x0, t, ycur, out, tcA, hA);

    int s = 0;
    for (int p = 0; p < 7; ++p) {            // N_POINTS-1 intervals
        for (int st = 0; st < 8; ++st, ++s) {// MAX_STEPS substeps
            ode_step_kernel<<<NBLK, 256, 0, stream>>>(tcA, hA, t, p, s,
                                                      ycur, y5, wsum,
                                                      WB1, WB2, WB3, WB4,
                                                      b1, b2, b3, b4);
            ode_commit_kernel<<<NBLK, 256, 0, stream>>>(tcA, hA, t, p, s,
                                                        (st == 7) ? 1 : 0,
                                                        ycur, y5, wsum, out);
        }
    }
}
